// ResMoELoRALinear_37778532336065
// MI455X (gfx1250) — compile-verified, matrix operand order checked by reading, not run on hardware
//
#include <hip/hip_runtime.h>
#include <hip/hip_bf16.h>

// ---------------- problem constants ----------------
#define DIN   2048
#define DOUT  2048
#define NTOK  8192      // B*S = 4*2048
#define NEXP  16
#define RDIM  64
#define CAP   8192      // worst-case tokens per expert (top-2 of softmax)
#define ROWU  (DIN / 2) // u32 (bf16 pair) per 2048-wide row
#define KT_N  (DIN / 32)

// ---------------- WMMA fragment types ----------------
typedef __attribute__((ext_vector_type(16))) __bf16   v16bf;
typedef __attribute__((ext_vector_type(8)))  float    v8f;
typedef __attribute__((ext_vector_type(8)))  unsigned v8u;
typedef __attribute__((ext_vector_type(4)))  unsigned v4u;

// float -> bf16 (round to nearest even), packed pair into u32
__device__ __forceinline__ unsigned f2bf_bits(float f) {
  unsigned u = __builtin_bit_cast(unsigned, f);
  return (u + 0x7FFFu + ((u >> 16) & 1u)) >> 16;
}
__device__ __forceinline__ unsigned pack_bf2(float a, float b) {
  return f2bf_bits(a) | (f2bf_bits(b) << 16);
}

// ---------------- CDNA5 async global->LDS staging ----------------
#if __has_builtin(__builtin_amdgcn_global_load_async_to_lds_b128)
#define ASYNC_LDS_OK 1
#else
#define ASYNC_LDS_OK 0
#endif

typedef int v4i_gcc __attribute__((vector_size(16)));
typedef __attribute__((address_space(1))) v4i_gcc* g_v4i_p;
typedef __attribute__((address_space(3))) v4i_gcc* l_v4i_p;

// Copy 16 bytes global -> LDS (ASYNCcnt-tracked, no VGPR round trip).
__device__ __forceinline__ void async_copy16(const unsigned* __restrict__ g,
                                             unsigned* __restrict__ l) {
#if ASYNC_LDS_OK
  __builtin_amdgcn_global_load_async_to_lds_b128(
      (g_v4i_p)(unsigned long long)(uintptr_t)g,
      (l_v4i_p)(unsigned)(uintptr_t)l, 0, 0);
#else
  *(uint4*)l = *(const uint4*)g;
#endif
}

__device__ __forceinline__ void wait_async0() {
#if ASYNC_LDS_OK
#if __has_builtin(__builtin_amdgcn_s_wait_asynccnt)
  __builtin_amdgcn_s_wait_asynccnt(0);
#else
  asm volatile("s_wait_asynccnt 0x0" ::: "memory");
#endif
#endif
}

__device__ __forceinline__ void wait_tensor0() {
#if __has_builtin(__builtin_amdgcn_s_wait_tensorcnt)
  __builtin_amdgcn_s_wait_tensorcnt(0);
#else
  asm volatile("s_wait_tensorcnt 0x0" ::: "memory");
#endif
}

// ---------------- CDNA5 Tensor Data Mover (TDM) ----------------
// Issue TENSOR_LOAD_TO_LDS for a 2D tile: tile_dim1 rows x 16 dwords (64B),
// source rows `1024` dwords apart (bf16 2048-wide matrix).  D# built per ISA
// CDNA5 ISA; 2-group form (<=2D tensor), one issue per WG.
__device__ __forceinline__ void tdm_load_tile_w(unsigned lds_byte_addr,
                                                const unsigned* __restrict__ gsrc,
                                                unsigned rows) {
  const unsigned long long ga = (unsigned long long)(uintptr_t)gsrc;
  v4u g0;
  g0[0] = 1u;                               // count=1, user descriptor
  g0[1] = lds_byte_addr;                    // lds_addr [63:32]
  g0[2] = (unsigned)ga;                     // global_addr lo
  g0[3] = (unsigned)(ga >> 32) | (2u << 30); // global_addr hi | type=2 (image)
  v8u g1;
  g1[0] = 2u << 16;                         // workgroup_mask=0, data_size=4B
  g1[1] = (1024u & 0xFFFFu) << 16;          // tensor_dim0[15:0] @ bits 63:48
  g1[2] = (rows & 0xFFFFu) << 16;           // tensor_dim0[31:16]=0 | tensor_dim1[15:0]
  g1[3] = 16u << 16;                        // tensor_dim1[31:16]=0 | tile_dim0=16
  g1[4] = 128u;                             // tile_dim1=128, tile_dim2=0
  g1[5] = 1024u;                            // tensor_dim0_stride[31:0]
  g1[6] = 0u;                               // stride[47:32]=0, dim1_stride lo=0
  g1[7] = 0u;                               // dim1_stride hi=0
  asm volatile("tensor_load_to_lds %0, %1" :: "s"(g0), "s"(g1) : "memory");
}

// Load one 16x32 bf16 operand fragment (A-layout; also valid for the B operand
// supplied as B^T stored N-major).  Per ISA 7.12.2: lanes 0-15 take bf16
// k[0..7] and k[16..23] of their row, lanes 16-31 take k[8..15] and k[24..31].
__device__ __forceinline__ v16bf load_frag_u32(const unsigned* __restrict__ p_row0,
                                               int stride_u32, int lane) {
  const int half = lane >> 4;
  const int r    = lane & 15;
  const unsigned* p = p_row0 + (size_t)r * stride_u32;
  v8u u;
  u[0] = p[half * 4 + 0];
  u[1] = p[half * 4 + 1];
  u[2] = p[half * 4 + 2];
  u[3] = p[half * 4 + 3];
  u[4] = p[8 + half * 4 + 0];
  u[5] = p[8 + half * 4 + 1];
  u[6] = p[8 + half * 4 + 2];
  u[7] = p[8 + half * 4 + 3];
  return __builtin_bit_cast(v16bf, u);
}

#define WMMA_BF16(a, b, c) \
  __builtin_amdgcn_wmma_f32_16x16x32_bf16(false, (a), false, (b), (short)0, (c), false, false)

// ---------------------------------------------------------------------------
// Kernel P: generic f32 -> packed-bf16 converter (memory bound, one pass)
// ---------------------------------------------------------------------------
__global__ __launch_bounds__(256) void k_cvt(const float* __restrict__ src,
                                             unsigned* __restrict__ dst, int n4) {
  int i = blockIdx.x * 256 + threadIdx.x;
  if (i >= n4) return;
  const float4 v = *(const float4*)(src + (size_t)i * 4);
  dst[(size_t)i * 2 + 0] = pack_bf2(v.x, v.y);
  dst[(size_t)i * 2 + 1] = pack_bf2(v.z, v.w);
}

// ---------------------------------------------------------------------------
// Kernel 1: rhl[t, 0:64] = x @ A^T ; rhl[t, 64:80] = x @ W_router^T  (f32 out)
// ---------------------------------------------------------------------------
__global__ __launch_bounds__(256) void k_rh_router(const unsigned* __restrict__ xb,
                                                   const unsigned* __restrict__ awb,
                                                   float* __restrict__ rhl) {
  __shared__ unsigned ldsX[2][128 * 16];
  __shared__ unsigned ldsA[2][80 * 16];
  const int tid  = threadIdx.x;
  const int lane = tid & 31;
  const int w    = tid >> 5;
  const int m0   = blockIdx.x * 128;

  auto stage = [&](int buf, int k0) {
    const int ku = k0 >> 1;
    #pragma unroll
    for (int i = 0; i < 2; ++i) {             // x tile: 512 x 16B chunks
      int c = tid + 256 * i;
      int row = c >> 2, q = c & 3;
      async_copy16(xb + (size_t)(m0 + row) * ROWU + ku + q * 4,
                   &ldsX[buf][row * 16 + q * 4]);
    }
    #pragma unroll
    for (int i = 0; i < 2; ++i) {             // [A;Wr] tile: 320 x 16B chunks
      int c = tid + 256 * i;
      if (c < 320) {
        int row = c >> 2, q = c & 3;
        async_copy16(awb + (size_t)row * ROWU + ku + q * 4,
                     &ldsA[buf][row * 16 + q * 4]);
      }
    }
  };

  v8f acc[5] = {};
  stage(0, 0);
  for (int kt = 0; kt < KT_N; ++kt) {
    wait_async0();
    __syncthreads();
    if (kt + 1 < KT_N) stage((kt + 1) & 1, (kt + 1) * 32);
    const unsigned* bx = ldsX[kt & 1];
    const unsigned* ba = ldsA[kt & 1];
    v16bf a = load_frag_u32(bx + (size_t)(w * 16) * 16, 16, lane);
    #pragma unroll
    for (int j = 0; j < 5; ++j) {
      v16bf b = load_frag_u32(ba + (size_t)(j * 16) * 16, 16, lane);
      acc[j] = WMMA_BF16(a, b, acc[j]);
    }
  }

  const int half = lane >> 4;
  const int nl   = lane & 15;
  #pragma unroll
  for (int j = 0; j < 5; ++j) {
    int col = j * 16 + nl;
    #pragma unroll
    for (int r = 0; r < 8; ++r) {
      int row = m0 + w * 16 + half * 8 + r;
      rhl[(size_t)row * 80 + col] = acc[j][r];
    }
  }
}

// ---------------------------------------------------------------------------
// Kernel 2: per-token softmax over 16 logits, top-2, renormalize; scatter
// w * res_hidden rows (bf16) into per-expert buckets (WMMA A row-major).
// ---------------------------------------------------------------------------
__global__ __launch_bounds__(256) void k_route(const float* __restrict__ rhl,
                                               unsigned* __restrict__ cnt,
                                               int* __restrict__ tok,
                                               unsigned* __restrict__ ag) {
  int t = blockIdx.x * 256 + threadIdx.x;
  if (t >= NTOK) return;
  const float* lg = rhl + (size_t)t * 80 + RDIM;
  float l[NEXP], p[NEXP];
  float m = -1e30f;
  #pragma unroll
  for (int i = 0; i < NEXP; ++i) { l[i] = lg[i]; m = fmaxf(m, l[i]); }
  float Z = 0.f;
  #pragma unroll
  for (int i = 0; i < NEXP; ++i) { p[i] = __expf(l[i] - m); Z += p[i]; }
  int i1 = 0;
  #pragma unroll
  for (int i = 1; i < NEXP; ++i) if (p[i] > p[i1]) i1 = i;
  int i2 = (i1 == 0) ? 1 : 0;
  #pragma unroll
  for (int i = 0; i < NEXP; ++i) if (i != i2 && i != i1 && p[i] > p[i2]) i2 = i;
  const float invZ  = 1.f / Z;
  const float denom = (p[i1] + p[i2]) * invZ + 1e-6f;
  const float wv[2] = { (p[i1] * invZ) / denom, (p[i2] * invZ) / denom };
  const int   ex[2] = { i1, i2 };
  const float* rh = rhl + (size_t)t * 80;
  #pragma unroll
  for (int j = 0; j < 2; ++j) {
    const int   e   = ex[j];
    const float wgt = wv[j];
    unsigned pos = atomicAdd(&cnt[e], 1u);
    tok[(size_t)e * CAP + pos] = t;
    unsigned* dst = ag + ((size_t)e * CAP + pos) * 32;   // 64 bf16 = 32 u32
    #pragma unroll
    for (int r = 0; r < 32; ++r)
      dst[r] = pack_bf2(wgt * rh[2 * r], wgt * rh[2 * r + 1]);
  }
}

// ---------------------------------------------------------------------------
// Kernel 0: out = x @ W_base^T + b_base.  128x128 tile, bf16 operands.
// x tile staged by the async-LDS engine (ASYNCcnt); W tile staged by the
// Tensor Data Mover (TENSORcnt), one tensor_load_to_lds per WG per K step.
// Double buffered; 8 waves (4x2), wave = 32x64.
// ---------------------------------------------------------------------------
__global__ __launch_bounds__(256) void k_base_gemm(const unsigned* __restrict__ xb,
                                                   const unsigned* __restrict__ wb,
                                                   const float* __restrict__ bias,
                                                   float* __restrict__ out) {
  __shared__ unsigned ldsX[2][128 * 16];
  __shared__ unsigned ldsW[2][128 * 16];
  const int tid  = threadIdx.x;
  const int lane = tid & 31;
  const int w    = tid >> 5;
  const int wm   = w & 3;
  const int wn   = w >> 2;
  const int m0   = blockIdx.x * 128;
  const int n0   = blockIdx.y * 128;

  auto stage = [&](int buf, int k0) {
    const int ku = k0 >> 1;
    // x tile: per-lane async copies (ASYNCcnt path)
    #pragma unroll
    for (int i = 0; i < 2; ++i) {
      int c = tid + 256 * i;
      int row = c >> 2, q = c & 3;
      async_copy16(xb + (size_t)(m0 + row) * ROWU + ku + q * 4,
                   &ldsX[buf][row * 16 + q * 4]);
    }
    // W tile: single TDM 2D-tile DMA, issued by wave 0 only (TENSORcnt path)
    if (w == 0) {
      tdm_load_tile_w((unsigned)(uintptr_t)&ldsW[buf][0],
                      wb + (size_t)n0 * ROWU + ku, 128u);
    }
  };

  v8f acc[2][4] = {};
  stage(0, 0);
  for (int kt = 0; kt < KT_N; ++kt) {
    wait_async0();
    wait_tensor0();
    __syncthreads();
    if (kt + 1 < KT_N) stage((kt + 1) & 1, (kt + 1) * 32);
    const unsigned* bx = ldsX[kt & 1];
    const unsigned* bw = ldsW[kt & 1];
    v16bf a0 = load_frag_u32(bx + (size_t)(wm * 32 + 0)  * 16, 16, lane);
    v16bf a1 = load_frag_u32(bx + (size_t)(wm * 32 + 16) * 16, 16, lane);
    #pragma unroll
    for (int j = 0; j < 4; ++j) {
      v16bf b = load_frag_u32(bw + (size_t)(wn * 64 + j * 16) * 16, 16, lane);
      acc[0][j] = WMMA_BF16(a0, b, acc[0][j]);
      acc[1][j] = WMMA_BF16(a1, b, acc[1][j]);
    }
  }

  const int half = lane >> 4;
  const int nl   = lane & 15;
  #pragma unroll
  for (int i = 0; i < 2; ++i) {
    #pragma unroll
    for (int j = 0; j < 4; ++j) {
      int col = n0 + wn * 64 + j * 16 + nl;
      float bv = bias[col];
      int rowb = m0 + wm * 32 + i * 16 + half * 8;
      #pragma unroll
      for (int r = 0; r < 8; ++r)
        out[(size_t)(rowb + r) * DOUT + col] = acc[i][j][r] + bv;
    }
  }
}

// ---------------------------------------------------------------------------
// Kernel 3: grouped expert delta GEMM.  grid = (mtile, ntile, expert).
// ---------------------------------------------------------------------------
__global__ __launch_bounds__(256) void k_delta(const unsigned* __restrict__ ag,
                                               const int* __restrict__ tok,
                                               const unsigned* __restrict__ cnt,
                                               const unsigned* __restrict__ Bb,
                                               float* __restrict__ out) {
  const int e     = blockIdx.z;
  const int cnt_e = (int)cnt[e];
  const int mbase = blockIdx.x * 16;
  if (mbase >= cnt_e) return;
  const int tid  = threadIdx.x;
  const int lane = tid & 31;
  const int w    = tid >> 5;
  const int n0   = blockIdx.y * 1024 + w * 128;

  const unsigned* arow = ag + ((size_t)e * CAP + mbase) * 32;
  v16bf af0 = load_frag_u32(arow, 32, lane);        // K 0..31
  v16bf af1 = load_frag_u32(arow + 16, 32, lane);   // K 32..63

  v8f acc[8] = {};
  #pragma unroll
  for (int j = 0; j < 8; ++j) {
    const int d0 = n0 + j * 16;
    const unsigned* brow = Bb + ((size_t)e * DOUT + d0) * 32;
    v16bf b0 = load_frag_u32(brow, 32, lane);
    v16bf b1 = load_frag_u32(brow + 16, 32, lane);
    acc[j] = WMMA_BF16(af0, b0, acc[j]);
    acc[j] = WMMA_BF16(af1, b1, acc[j]);
  }

  const int half = lane >> 4;
  const int nl   = lane & 15;
  #pragma unroll
  for (int j = 0; j < 8; ++j) {
    #pragma unroll
    for (int r = 0; r < 8; ++r) {
      int pos = mbase + half * 8 + r;
      if (pos < cnt_e) {
        int t   = tok[(size_t)e * CAP + pos];
        int col = n0 + j * 16 + nl;
        __hip_atomic_fetch_add(out + (size_t)t * DOUT + col, acc[j][r],
                               __ATOMIC_RELAXED, __HIP_MEMORY_SCOPE_AGENT);
      }
    }
  }
}

// ---------------------------------------------------------------------------
extern "C" void kernel_launch(void* const* d_in, const int* in_sizes, int n_in,
                              void* d_out, int out_size, void* d_ws, size_t ws_size,
                              hipStream_t stream) {
  (void)in_sizes; (void)n_in; (void)out_size; (void)ws_size;
  const float* x  = (const float*)d_in[0];   // [4,2048,2048]
  const float* Wb = (const float*)d_in[1];   // [2048,2048]
  const float* bb = (const float*)d_in[2];   // [2048]
  const float* A  = (const float*)d_in[3];   // [64,2048]
  const float* Bm = (const float*)d_in[4];   // [16,2048,64]
  const float* Wr = (const float*)d_in[5];   // [16,2048]
  // d_in[6] = top_k (device scalar) — fixed at 2 for this shape set.
  float* out = (float*)d_out;

  char* ws = (char*)d_ws;
  size_t off = 0;
  unsigned* xb   = (unsigned*)(ws + off); off += (size_t)NTOK * DIN * 2;          // 33.6 MB
  unsigned* wbb  = (unsigned*)(ws + off); off += (size_t)DOUT * DIN * 2;          //  8.4 MB
  unsigned* awb  = (unsigned*)(ws + off); off += (size_t)(RDIM + NEXP) * DIN * 2; //  0.33 MB
  float*    rhl  = (float*)(ws + off);    off += (size_t)NTOK * 80 * 4;           //  2.62 MB
  unsigned* Bb16 = (unsigned*)(ws + off); off += (size_t)NEXP * DOUT * RDIM * 2;  //  4.19 MB
  unsigned* cnt  = (unsigned*)(ws + off); off += 256;
  int*      tok  = (int*)(ws + off);      off += (size_t)NEXP * CAP * 4;          //  0.52 MB
  unsigned* ag   = (unsigned*)(ws + off); off += (size_t)NEXP * CAP * RDIM * 2;   // 16.8 MB

  (void)hipMemsetAsync(cnt, 0, 256, stream);

  // precision conversion prep (memory bound)
  k_cvt<<<dim3(NTOK * DIN / 4 / 256), dim3(256), 0, stream>>>(x, xb, NTOK * DIN / 4);
  k_cvt<<<dim3(DOUT * DIN / 4 / 256), dim3(256), 0, stream>>>(Wb, wbb, DOUT * DIN / 4);
  k_cvt<<<dim3(RDIM * DIN / 4 / 256), dim3(256), 0, stream>>>(A, awb, RDIM * DIN / 4);
  k_cvt<<<dim3(NEXP * DIN / 4 / 256), dim3(256), 0, stream>>>(
      Wr, awb + (size_t)RDIM * DIN / 2, NEXP * DIN / 4);
  k_cvt<<<dim3(NEXP * DOUT * RDIM / 4 / 256), dim3(256), 0, stream>>>(
      Bm, Bb16, NEXP * DOUT * RDIM / 4);

  k_rh_router<<<dim3(NTOK / 128), dim3(256), 0, stream>>>(xb, awb, rhl);
  k_route<<<dim3(NTOK / 256), dim3(256), 0, stream>>>(rhl, cnt, tok, ag);
  k_base_gemm<<<dim3(NTOK / 128, DOUT / 128), dim3(256), 0, stream>>>(xb, wbb, bb, out);
  k_delta<<<dim3(CAP / 16, DOUT / 1024, NEXP), dim3(256), 0, stream>>>(
      ag, tok, cnt, Bb16, out);
}
